// LSTMModel_14053132992728
// MI455X (gfx1250) — matrix operand order repaired and checked by reading, not compiled, not run on hardware
//
#include <hip/hip_runtime.h>
#include <stdint.h>

#define HID     64
#define NLAYER  4
#define BT      16            // batch rows per workgroup -> 64 persistent WGs across the chip
#define S_LEN   512
#define B_TOT   1024
#define GATES   256           // 4*HID
#define KP0     96            // layer 0: fan_in 65 padded to 3 K-steps of 32 (K permuted: h first, x at k=64)
#define KP      128           // layers 1..3: fan_in 128 = 4 K-steps
#define LN_EPS  1e-5f

typedef __attribute__((ext_vector_type(16))) __bf16 v16bf;
typedef __attribute__((ext_vector_type(8)))  __bf16 v8bf;
typedef __attribute__((ext_vector_type(4)))  __bf16 v4bf;
typedef __attribute__((ext_vector_type(8)))  float  v8f;
typedef __attribute__((ext_vector_type(4)))  float  v4f;

// ---- LDS layout (element offsets into the bf16 region) ----
constexpr int WT_SZ0  = GATES * KP0;            // 24576
constexpr int WT_SZ   = GATES * KP;             // 32768
constexpr int WT_OFF0 = 0;
constexpr int WT_OFF1 = WT_OFF0 + WT_SZ0;
constexpr int WT_OFF2 = WT_OFF1 + WT_SZ;
constexpr int WT_OFF3 = WT_OFF2 + WT_SZ;
constexpr int A_OFF0  = WT_OFF3 + WT_SZ;        // A-matrix staging per layer
constexpr int A_OFF1  = A_OFF0 + BT * KP0;
constexpr int A_OFF2  = A_OFF1 + BT * KP;
constexpr int A_OFF3  = A_OFF2 + BT * KP;
constexpr int BF_TOTAL = A_OFF3 + BT * KP;

// f32 region sits after the bf16 region
constexpr int F32_Z    = 0;                     // zbuf [BT][GATES]
constexpr int F32_BIAS = F32_Z + BT * GATES;    // bias [4][GATES]
constexpr int F32_PRM  = F32_BIAS + NLAYER * GATES; // ln_g[64] ln_b[64] wout[64] bout
constexpr int F32_TOTAL = F32_PRM + 256;

constexpr size_t SMEM_BYTES = (size_t)BF_TOTAL * 2 + (size_t)F32_TOTAL * 4;
static_assert(SMEM_BYTES <= 320u * 1024u, "LDS budget exceeded");

__device__ __forceinline__ __bf16 f2bf(float f) {
  union { float f; uint32_t u; } v; v.f = f;
  uint32_t r = v.u + 0x7FFFu + ((v.u >> 16) & 1u);   // round-to-nearest-even
  uint16_t h = (uint16_t)(r >> 16);
  return __builtin_bit_cast(__bf16, h);
}

__device__ __forceinline__ float sigf(float x) {
  return 1.0f / (1.0f + __expf(-x));
}
__device__ __forceinline__ float tanhf_(float x) {
  float ax = fabsf(x);
  float e  = __expf(-2.0f * ax);
  float r  = (1.0f - e) / (1.0f + e);
  return copysignf(r, x);
}

// ---- load a wave's B (weight) fragments from LDS into registers, once ----
template<int KSTEPS>
__device__ __forceinline__ void load_wfrags(const __bf16* __restrict__ wt, int kp,
                                            int wave, int lane, v16bf (&bw)[KSTEPS][2]) {
  const int lhalf = (lane >> 4) & 1;
  const int l15   = lane & 15;
  const int nbase = wave * 32;
#pragma unroll
  for (int ks = 0; ks < KSTEPS; ++ks)
#pragma unroll
    for (int ni = 0; ni < 2; ++ni)
      bw[ks][ni] = *(const v16bf*)(wt + (nbase + ni * 16 + l15) * kp + ks * 32 + lhalf * 16);
}

// One layer GEMM with register-resident weights:
// z[BT(=16) x 256] = A[16 x kp] * Wt^T + bias via bf16 WMMA; wave owns 32 columns (2 N-tiles, 1 M-tile).
template<int KSTEPS>
__device__ __forceinline__ void gemm_z_reg(const v16bf (&bw)[KSTEPS][2],
                                           const __bf16* __restrict__ ab,   // [BT][kp]
                                           const float*  __restrict__ biasrow,
                                           float*        __restrict__ zb,   // [BT][GATES]
                                           int kp, int wave, int lane) {
  const int lhalf = (lane >> 4) & 1;
  const int l15   = lane & 15;
  const int nbase = wave * 32;

  v8f acc[2];
#pragma unroll
  for (int ni = 0; ni < 2; ++ni)
#pragma unroll
    for (int r = 0; r < 8; ++r) acc[ni][r] = 0.0f;

#pragma unroll
  for (int ks = 0; ks < KSTEPS; ++ks) {
    // A fragment (16x32 bf16): lane<16 holds K {0..15}, lane>=16 holds K {16..31}
    const __bf16* ap = ab + l15 * kp + ks * 32 + lhalf * 16;
    v8bf lo = *(const v8bf*)(ap);
    v8bf hi = *(const v8bf*)(ap + 8);
    v16bf afr;
#pragma unroll
    for (int i = 0; i < 8; ++i) { afr[i] = lo[i]; afr[i + 8] = hi[i]; }
#pragma unroll
    for (int ni = 0; ni < 2; ++ni)
      acc[ni] = __builtin_amdgcn_wmma_f32_16x16x32_bf16(
          false, afr, false, bw[ks][ni], (short)0, acc[ni], false, false);
  }

  // C/D layout: VGPR r -> M = r + 8*lhalf; N = l15
#pragma unroll
  for (int ni = 0; ni < 2; ++ni) {
    const int col = nbase + ni * 16 + l15;
    const float bv = biasrow[col];
#pragma unroll
    for (int r = 0; r < 8; ++r) {
      const int row = lhalf * 8 + r;
      zb[row * GATES + col] = acc[ni][r] + bv;
    }
  }
}

// Elementwise LSTM cell for 4 contiguous hidden units of one batch row (vectorized).
// Returns the 4 fresh h values in hreg for optional further fusion (top-layer LN).
__device__ __forceinline__ void cell_update(const float* __restrict__ zr,     // z row [GATES]
                                            float*       __restrict__ cr,     // c state [4] in regs
                                            __bf16*      __restrict__ ahSelf, // h dest row (own layer, 8B aligned)
                                            __bf16*      __restrict__ anNext, // h dest row (next layer input) or null
                                            float*       __restrict__ hreg,   // [4] out
                                            int ej0) {
  v4f zi = *(const v4f*)(zr + 0 * HID + ej0);
  v4f zf = *(const v4f*)(zr + 1 * HID + ej0);
  v4f zg = *(const v4f*)(zr + 2 * HID + ej0);
  v4f zo = *(const v4f*)(zr + 3 * HID + ej0);
  v4bf hv;
#pragma unroll
  for (int r = 0; r < 4; ++r) {
    float iv = sigf(zi[r]);
    float fv = sigf(zf[r]);
    float gv = tanhf_(zg[r]);
    float ov = sigf(zo[r]);
    float c  = fv * cr[r] + iv * gv;
    cr[r] = c;
    float h = ov * tanhf_(c);
    hv[r]   = f2bf(h);
    hreg[r] = h;
  }
  *(v4bf*)(ahSelf + ej0) = hv;                       // single 8B LDS store
  if (anNext) *(v4bf*)(anNext + ej0) = hv;
}

// butterfly sum across the 16 lanes owning one batch row (masks 1,2,4,8 within wave32)
__device__ __forceinline__ float rowsum16(float v) {
  v += __shfl_xor(v, 1, 32);
  v += __shfl_xor(v, 2, 32);
  v += __shfl_xor(v, 4, 32);
  v += __shfl_xor(v, 8, 32);
  return v;
}

__global__ void __launch_bounds__(256)
lstm_fused_kernel(const float* __restrict__ x,
                  const float* __restrict__ gW0, const float* __restrict__ gb0,
                  const float* __restrict__ gW1, const float* __restrict__ gb1,
                  const float* __restrict__ gW2, const float* __restrict__ gb2,
                  const float* __restrict__ gW3, const float* __restrict__ gb3,
                  const float* __restrict__ ln_g, const float* __restrict__ ln_b,
                  const float* __restrict__ wout, const float* __restrict__ bout,
                  float* __restrict__ out) {
  extern __shared__ char smem[];
  __bf16* bfp = (__bf16*)smem;
  float*  fp  = (float*)(smem + (size_t)BF_TOTAL * 2);

  float* zbuf = fp + F32_Z;
  float* bias = fp + F32_BIAS;
  float* prm  = fp + F32_PRM;

  __bf16* wt[NLAYER]   = { bfp + WT_OFF0, bfp + WT_OFF1, bfp + WT_OFF2, bfp + WT_OFF3 };
  __bf16* abuf[NLAYER] = { bfp + A_OFF0,  bfp + A_OFF1,  bfp + A_OFF2,  bfp + A_OFF3  };
  const float* gWs[NLAYER] = { gW0, gW1, gW2, gW3 };
  const float* gbs[NLAYER] = { gb0, gb1, gb2, gb3 };

  const int tid = threadIdx.x;

  // ---- one-time: stage weights (transposed, bf16, zero-padded K), biases, params; zero A buffers ----
  // Layer 0 K-permutation: k in [0,64) <- W rows 1..64 (h weights); k==64 <- W row 0 (x weight); k>64 zero.
  for (int l = 0; l < NLAYER; ++l) {
    const int kp = (l == 0) ? KP0 : KP;
    const float* W = gWs[l];
    __bf16* dst = wt[l];
    for (int idx = tid; idx < GATES * kp; idx += blockDim.x) {
      const int n = idx / kp;
      const int k = idx - n * kp;
      float w;
      if (l == 0) {
        if (k < HID)       w = W[(k + 1) * GATES + n];
        else if (k == HID) w = W[0 * GATES + n];
        else               w = 0.0f;
      } else {
        w = (k < 2 * HID) ? W[k * GATES + n] : 0.0f;
      }
      dst[idx] = f2bf(w);
    }
    for (int n = tid; n < GATES; n += blockDim.x) bias[l * GATES + n] = gbs[l][n];
  }
  for (int idx = tid; idx < (BF_TOTAL - A_OFF0); idx += blockDim.x)
    bfp[A_OFF0 + idx] = f2bf(0.0f);
  if (tid < HID) {
    prm[tid]       = ln_g[tid];
    prm[64 + tid]  = ln_b[tid];
    prm[128 + tid] = wout[tid];
  }
  if (tid == 0) prm[192] = bout[0];
  __syncthreads();

  const int wave = tid >> 5;
  const int lane = tid & 31;
  const int eb   = tid >> 4;            // batch row for elementwise phase (16 lanes per row)
  const int ej0  = (tid & 15) * 4;      // hidden-unit base for elementwise phase
  const int b0g  = blockIdx.x * BT;

  // ---- hoist this wave's weight fragments into registers (invariant over all 512 steps) ----
  v16bf bw0[3][2], bw1[4][2], bw2[4][2], bw3[4][2];
  load_wfrags<3>(wt[0], KP0, wave, lane, bw0);
  load_wfrags<4>(wt[1], KP,  wave, lane, bw1);
  load_wfrags<4>(wt[2], KP,  wave, lane, bw2);
  load_wfrags<4>(wt[3], KP,  wave, lane, bw3);

  // ---- loop-invariant LN/projection constants (per thread: 4 hidden units of row eb) ----
  float gwr[4];                          // ln_g[j] * wout[j]
  float bwp = 0.f;                       // partial sum of ln_b[j] * wout[j]
#pragma unroll
  for (int r = 0; r < 4; ++r) {
    const int j = ej0 + r;
    gwr[r] = prm[j] * prm[128 + j];
    bwp   += prm[64 + j] * prm[128 + j];
  }
  const float ycst = rowsum16(bwp) + prm[192];   // sum(ln_b*wout) + bout, same for all rows

  float c0[4] = {0.f}, c1[4] = {0.f}, c2[4] = {0.f}, c3[4] = {0.f};
  float hreg[4];

  // stage x(0); all later x(t+1) stores are folded into the cell-l0 phase
  if (tid < BT)
    abuf[0][tid * KP0 + HID] = f2bf(x[(size_t)(b0g + tid) * S_LEN]);
  __syncthreads();

  for (int t = 0; t < S_LEN; ++t) {
    // layer 0 (h at K 0..63, so self-h store is aligned at col 0)
    gemm_z_reg<3>(bw0, abuf[0], bias + 0 * GATES, zbuf, KP0, wave, lane);
    __syncthreads();
    cell_update(zbuf + eb * GATES, c0, abuf[0] + eb * KP0, abuf[1] + eb * KP, hreg, ej0);
    if (tid < BT && t + 1 < S_LEN) {    // stage x(t+1); reader is gemm-l0 of t+1, many barriers away
      abuf[0][tid * KP0 + HID] = f2bf(x[(size_t)(b0g + tid) * S_LEN + t + 1]);
      if (t + 16 < S_LEN)
        __builtin_prefetch(x + (size_t)(b0g + tid) * S_LEN + t + 16, 0, 0);
    }
    __syncthreads();
    // layer 1
    gemm_z_reg<4>(bw1, abuf[1], bias + 1 * GATES, zbuf, KP, wave, lane);
    __syncthreads();
    cell_update(zbuf + eb * GATES, c1, abuf[1] + eb * KP + HID, abuf[2] + eb * KP, hreg, ej0);
    __syncthreads();
    // layer 2
    gemm_z_reg<4>(bw2, abuf[2], bias + 2 * GATES, zbuf, KP, wave, lane);
    __syncthreads();
    cell_update(zbuf + eb * GATES, c2, abuf[2] + eb * KP + HID, abuf[3] + eb * KP, hreg, ej0);
    __syncthreads();
    // layer 3 (top)
    gemm_z_reg<4>(bw3, abuf[3], bias + 3 * GATES, zbuf, KP, wave, lane);
    __syncthreads();
    cell_update(zbuf + eb * GATES, c3, abuf[3] + eb * KP + HID, nullptr, hreg, ej0);

    // fused LayerNorm + 64->1 projection via intra-wave butterflies (16 lanes per batch row)
    {
      float s1 = 0.f, s2 = 0.f;
#pragma unroll
      for (int r = 0; r < 4; ++r) { s1 += hreg[r]; s2 += hreg[r] * hreg[r]; }
      s1 = rowsum16(s1);
      s2 = rowsum16(s2);
      const float mu   = s1 * (1.0f / HID);
      const float var  = s2 * (1.0f / HID) - mu * mu;
      const float rstd = rsqrtf(var + LN_EPS);
      float p = 0.f;
#pragma unroll
      for (int r = 0; r < 4; ++r) p += (hreg[r] - mu) * gwr[r];
      p = rowsum16(p);
      if ((tid & 15) == 0)
        out[(size_t)(b0g + eb) * S_LEN + t] = rstd * p + ycst;
    }
    __syncthreads();   // zbuf WAR: cell-l3 readers vs gemm-l0 writers of t+1
  }
}

extern "C" void kernel_launch(void* const* d_in, const int* in_sizes, int n_in,
                              void* d_out, int out_size, void* d_ws, size_t ws_size,
                              hipStream_t stream) {
  (void)in_sizes; (void)n_in; (void)out_size; (void)d_ws; (void)ws_size;
  const float* x    = (const float*)d_in[0];
  const float* W0   = (const float*)d_in[1];
  const float* b0   = (const float*)d_in[2];
  const float* W1   = (const float*)d_in[3];
  const float* b1   = (const float*)d_in[4];
  const float* W2   = (const float*)d_in[5];
  const float* b2   = (const float*)d_in[6];
  const float* W3   = (const float*)d_in[7];
  const float* b3   = (const float*)d_in[8];
  const float* ln_g = (const float*)d_in[9];
  const float* ln_b = (const float*)d_in[10];
  const float* Wout = (const float*)d_in[11];
  const float* bout = (const float*)d_in[12];
  float* out = (float*)d_out;

  (void)hipFuncSetAttribute((const void*)lstm_fused_kernel,
                            hipFuncAttributeMaxDynamicSharedMemorySize,
                            (int)SMEM_BYTES);
  lstm_fused_kernel<<<B_TOT / BT, 256, SMEM_BYTES, stream>>>(
      x, W0, b0, W1, b1, W2, b2, W3, b3, ln_g, ln_b, Wout, bout, out);
}
